// DenseMOE_16509854286322
// MI455X (gfx1250) — matrix operand order repaired and checked by reading, not compiled, not run on hardware
//
#include <hip/hip_runtime.h>

// Problem constants (from reference): B=4, S=2048, D=1024, E=8, H=4096, top_k=2
#define NTOK 8192
#define D_   1024
#define E_   8
#define H_   4096
#define CAP  8192   // per-(rank,expert) list capacity

typedef __attribute__((ext_vector_type(16))) __bf16       v16bf;
typedef __attribute__((ext_vector_type(8)))  float        v8f;
typedef __attribute__((ext_vector_type(4)))  unsigned int v4u;
typedef __attribute__((ext_vector_type(4)))  int          v4i;

union BFrag { v16bf v; v4u u[2]; };

// ---- CDNA5 async global->LDS copy (ASYNCcnt-tracked), with sync fallback ----
#if __has_builtin(__builtin_amdgcn_global_load_async_to_lds_b128)
#define HAVE_ASYNC_LDS 1
#define GAS1(p) ((__attribute__((address_space(1))) v4i*)(p))
#define LAS3(p) ((__attribute__((address_space(3))) v4i*)(p))
#endif

__device__ __forceinline__ void cp16_g2l(const unsigned short* g, unsigned short* l) {
#ifdef HAVE_ASYNC_LDS
  __builtin_amdgcn_global_load_async_to_lds_b128(GAS1(g), LAS3(l), 0, 0);
#else
  *(v4u*)l = *(const v4u*)g;
#endif
}

__device__ __forceinline__ void stage_fence_barrier() {
#ifdef HAVE_ASYNC_LDS
#if __has_builtin(__builtin_amdgcn_s_wait_asynccnt)
  __builtin_amdgcn_s_wait_asynccnt(0);
#else
  asm volatile("s_wait_asynccnt 0x0" ::: "memory");
#endif
#endif
  __syncthreads();
}

__device__ __forceinline__ unsigned short f32_to_bf16(float f) {
  unsigned int u = __float_as_uint(f);
  u += 0x7FFFu + ((u >> 16) & 1u);          // round-to-nearest-even
  return (unsigned short)(u >> 16);
}
__device__ __forceinline__ unsigned int pack_bf16x2(float a, float b) {
  return (unsigned int)f32_to_bf16(a) | ((unsigned int)f32_to_bf16(b) << 16);
}

// ---------------------------------------------------------------- init
__global__ void init_counts(int* cnt) {
  if (threadIdx.x < 16) cnt[threadIdx.x] = 0;
}

// ---------------------------------------------------------------- router
// One wave32 per token: logits = x[tok] @ router_w^T + b; top-2 softmax;
// append (tok, gate) into per-rank per-expert lists.
__global__ __launch_bounds__(256) void router_kernel(
    const float* __restrict__ x, const float* __restrict__ rw,
    const float* __restrict__ rb, int* __restrict__ cnt,
    int* __restrict__ tokL, float* __restrict__ gateL) {
  const int lane = threadIdx.x & 31;
  const int wid  = threadIdx.x >> 5;
  const int tok  = blockIdx.x * 8 + wid;
  const float* xr = x + (size_t)tok * D_;

  float acc[E_];
#pragma unroll
  for (int e = 0; e < E_; ++e) acc[e] = 0.f;

  for (int d = lane; d < D_; d += 32) {
    float xv = xr[d];
#pragma unroll
    for (int e = 0; e < E_; ++e) acc[e] += xv * rw[e * D_ + d];
  }
#pragma unroll
  for (int e = 0; e < E_; ++e)
    for (int m = 16; m >= 1; m >>= 1) acc[e] += __shfl_xor(acc[e], m, 32);

  if (lane == 0) {
    float l[E_];
#pragma unroll
    for (int e = 0; e < E_; ++e) l[e] = acc[e] + rb[e];
    int e0 = 0;
#pragma unroll
    for (int e = 1; e < E_; ++e) if (l[e] > l[e0]) e0 = e;
    int e1 = (e0 == 0) ? 1 : 0;
#pragma unroll
    for (int e = 0; e < E_; ++e) if (e != e0 && l[e] > l[e1]) e1 = e;
    float mx = fmaxf(l[e0], l[e1]);
    float z0 = __expf(l[e0] - mx), z1 = __expf(l[e1] - mx);
    float inv = 1.f / (z0 + z1);
    int p0 = atomicAdd(&cnt[e0], 1);
    tokL[e0 * CAP + p0] = tok;  gateL[e0 * CAP + p0] = z0 * inv;
    int p1 = atomicAdd(&cnt[E_ + e1], 1);
    tokL[E_ * CAP + e1 * CAP + p1] = tok;  gateL[E_ * CAP + e1 * CAP + p1] = z1 * inv;
  }
}

// ---------------------------------------------------------------- f32 -> bf16 (flat, 8/thread)
__global__ __launch_bounds__(256) void cvt_f32_bf16(
    const float* __restrict__ src, unsigned short* __restrict__ dst) {
  size_t base = (size_t)(blockIdx.x * 256 + threadIdx.x) * 8;
  float4 a = *(const float4*)(src + base);
  float4 b = *(const float4*)(src + base + 4);
  v4u o;
  o.x = pack_bf16x2(a.x, a.y);
  o.y = pack_bf16x2(a.z, a.w);
  o.z = pack_bf16x2(b.x, b.y);
  o.w = pack_bf16x2(b.z, b.w);
  *(v4u*)(dst + base) = o;
}

// ---------------------------------------------------------------- transpose + convert
// src f32 [E][R][C] -> dst bf16 [E][C][R]  (weights become K-contiguous for GEMM B)
__global__ __launch_bounds__(256) void transpose_cvt(
    const float* __restrict__ src, unsigned short* __restrict__ dst, int R, int C) {
  __shared__ float t[32][33];
  const int e = blockIdx.z;
  const size_t sbase = (size_t)e * R * C;
  const int c0 = blockIdx.x * 32, r0 = blockIdx.y * 32;
#pragma unroll
  for (int i = 0; i < 4; ++i) {
    int r = r0 + threadIdx.y + i * 8;
    t[threadIdx.y + i * 8][threadIdx.x] = src[sbase + (size_t)r * C + c0 + threadIdx.x];
  }
  __syncthreads();
#pragma unroll
  for (int i = 0; i < 4; ++i) {
    int c = c0 + threadIdx.y + i * 8;
    dst[sbase + (size_t)c * R + r0 + threadIdx.x] = f32_to_bf16(t[threadIdx.x][threadIdx.y + i * 8]);
  }
}

// ---------------------------------------------------------------- fused MoE GEMM
// MODE 0: h = relu(x @ w1 + b1) -> bf16 Hout       (K=D, Nall=H)
// MODE 1: out  = gate * (h @ w2 + b2)              (K=H, Nall=D)  rank-0 pass
// MODE 2: out += gate * (h @ w2 + b2)              (K=H, Nall=D)  rank-1 pass
// Block: 128 gathered token rows x 128 output cols, 8 waves (2Mx4N),
// each wave owns 4x2 16x16 f32 accumulators; K in steps of 32, double-buffered
// LDS staged by async global->LDS copies (ASYNCcnt).
template <int MODE>
__global__ __launch_bounds__(256) void moe_ffn_gemm(
    const unsigned short* __restrict__ Abase,   // bf16 [NTOK][K] by token
    const unsigned short* __restrict__ Btbase,  // bf16 [E][Nall][K] (pre-transposed)
    const float* __restrict__ bias,             // f32 [E][Nall]
    const int*   __restrict__ cnts,             // [E]
    const int*   __restrict__ toks,             // [E][CAP]
    const float* __restrict__ gates,            // [E][CAP]
    unsigned short* __restrict__ Hout,          // MODE 0
    float*          __restrict__ Fout,          // MODE 1/2
    int K, int Nall) {
  __shared__ __align__(16) unsigned short As[2][128][40];
  __shared__ __align__(16) unsigned short Bs[2][128][40];
  __shared__ int   sTok[128];
  __shared__ float sGate[128];

  const int e   = blockIdx.z;
  const int cnt = cnts[e];
  const int m0  = blockIdx.x * 128;
  if (m0 >= cnt) return;                        // covers cnt == 0
  const int n0  = blockIdx.y * 128;

  const int tid  = threadIdx.x;
  const int lane = tid & 31;
  const int wid  = tid >> 5;

  if (tid < 128) {
    int p = m0 + tid;
    if (p > cnt - 1) p = cnt - 1;               // clamp padded rows (stores masked later)
    sTok[tid]  = toks[e * CAP + p];
    sGate[tid] = gates[e * CAP + p];
  }
  __syncthreads();

  // --- staging assignment: each thread moves 2 x 16B for A and for B per K-step
  const int rowS = tid >> 2;                    // 0..63
  const int segS = (tid & 3) * 8;               // element offset {0,8,16,24}
  const unsigned short* aSrc0 = Abase + (size_t)sTok[rowS]      * K + segS;
  const unsigned short* aSrc1 = Abase + (size_t)sTok[rowS + 64] * K + segS;
  const unsigned short* Bt    = Btbase + ((size_t)e * Nall + n0) * (size_t)K;
  const unsigned short* bSrc0 = Bt + (size_t)rowS        * K + segS;
  const unsigned short* bSrc1 = Bt + (size_t)(rowS + 64) * K + segS;
  unsigned short* aDst0 = &As[0][rowS][segS];
  unsigned short* aDst1 = &As[0][rowS + 64][segS];
  unsigned short* bDst0 = &Bs[0][rowS][segS];
  unsigned short* bDst1 = &Bs[0][rowS + 64][segS];
  const int BUFSTRIDE = 128 * 40;

  v8f zero = {};
  v8f acc[4][2];
#pragma unroll
  for (int mi = 0; mi < 4; ++mi)
#pragma unroll
    for (int ni = 0; ni < 2; ++ni) acc[mi][ni] = zero;

  const int wm   = (wid >> 2) * 64;             // wave M offset: 0 / 64
  const int wn   = (wid & 3) * 32;              // wave N offset: 0..96
  const int frow = lane & 15;                   // fragment row/col within tile
  const int aklo = (lane >> 4) * 8;             // A: split-K lane halves
  const int bklo = (lane >> 4) * 8;             // B: split-K lane halves

  const int nsteps = K >> 5;

  // prologue: stage step 0 into buffer 0
  cp16_g2l(aSrc0, aDst0);
  cp16_g2l(aSrc1, aDst1);
  cp16_g2l(bSrc0, bDst0);
  cp16_g2l(bSrc1, bDst1);
  stage_fence_barrier();

  for (int s = 0; s < nsteps; ++s) {
    const int buf = s & 1;
    if (s + 1 < nsteps) {
      const int k1 = (s + 1) << 5;
      const int d  = (buf ^ 1) * BUFSTRIDE;
      cp16_g2l(aSrc0 + k1, aDst0 + d);
      cp16_g2l(aSrc1 + k1, aDst1 + d);
      cp16_g2l(bSrc0 + k1, bDst0 + d);
      cp16_g2l(bSrc1 + k1, bDst1 + d);
      if (s + 3 < nsteps) {                     // global_prefetch_b8 on future weights
        __builtin_prefetch(bSrc0 + ((s + 3) << 5), 0, 0);
        __builtin_prefetch(bSrc1 + ((s + 3) << 5), 0, 0);
      }
    }

    BFrag af[4], bfr[2];
#pragma unroll
    for (int mi = 0; mi < 4; ++mi) {
      const unsigned short* p = &As[buf][wm + mi * 16 + frow][aklo];
      af[mi].u[0] = *(const v4u*)p;             // K {0..7} / {8..15}
      af[mi].u[1] = *(const v4u*)(p + 16);      // K {16..23} / {24..31}
    }
#pragma unroll
    for (int ni = 0; ni < 2; ++ni) {
      const unsigned short* p = &Bs[buf][wn + ni * 16 + frow][bklo];
      bfr[ni].u[0] = *(const v4u*)p;            // K {0..7} / {8..15}
      bfr[ni].u[1] = *(const v4u*)(p + 16);     // K {16..23} / {24..31}
    }
#pragma unroll
    for (int mi = 0; mi < 4; ++mi)
#pragma unroll
      for (int ni = 0; ni < 2; ++ni)
        acc[mi][ni] = __builtin_amdgcn_wmma_f32_16x16x32_bf16(
            false, af[mi].v, false, bfr[ni].v, (short)0, acc[mi][ni], false, false);

    stage_fence_barrier();
  }

  // --- epilogue: C layout -> lane holds (M = j + 8*(lane>>4), N = lane&15) per tile
  const int crow = (lane >> 4) * 8;
#pragma unroll
  for (int mi = 0; mi < 4; ++mi) {
#pragma unroll
    for (int ni = 0; ni < 2; ++ni) {
      const int col = n0 + wn + ni * 16 + (lane & 15);
      const float bv = bias[(size_t)e * Nall + col];
#pragma unroll
      for (int j = 0; j < 8; ++j) {
        const int mrow = wm + mi * 16 + crow + j;
        const int pos  = m0 + mrow;
        if (pos < cnt) {
          if constexpr (MODE == 0) {
            float v = acc[mi][ni][j] + bv;
            v = v > 0.f ? v : 0.f;              // ReLU
            Hout[(size_t)sTok[mrow] * Nall + col] = f32_to_bf16(v);
          } else {
            float v = (acc[mi][ni][j] + bv) * sGate[mrow];
            float* po = &Fout[(size_t)sTok[mrow] * Nall + col];
            if constexpr (MODE == 1) *po = v;   // rank-0: plain store (covers all tokens)
            else                     *po += v;  // rank-1: accumulate (token unique per rank)
          }
        }
      }
    }
  }
}

// ---------------------------------------------------------------- launch
extern "C" void kernel_launch(void* const* d_in, const int* in_sizes, int n_in,
                              void* d_out, int out_size, void* d_ws, size_t ws_size,
                              hipStream_t stream) {
  (void)in_sizes; (void)n_in; (void)out_size; (void)ws_size;
  const float* x        = (const float*)d_in[0];
  const float* router_w = (const float*)d_in[1];
  const float* router_b = (const float*)d_in[2];
  const float* w1       = (const float*)d_in[3];
  const float* b1       = (const float*)d_in[4];
  const float* w2       = (const float*)d_in[5];
  const float* b2       = (const float*)d_in[6];
  float* out = (float*)d_out;

  // workspace carve-up (requires ~219 MB)
  char* ws = (char*)d_ws;
  size_t off = 0;
  auto carve = [&](size_t bytes) -> char* {
    off = (off + 255) & ~(size_t)255;
    char* p = ws + off;
    off += bytes;
    return p;
  };
  int*            cnt   = (int*)carve(16 * sizeof(int));
  int*            tokL  = (int*)carve((size_t)2 * E_ * CAP * sizeof(int));
  float*          gateL = (float*)carve((size_t)2 * E_ * CAP * sizeof(float));
  unsigned short* xbf   = (unsigned short*)carve((size_t)NTOK * D_ * 2);
  unsigned short* w1t   = (unsigned short*)carve((size_t)E_ * H_ * D_ * 2); // [E][H][D]
  unsigned short* w2t   = (unsigned short*)carve((size_t)E_ * D_ * H_ * 2); // [E][D][H]
  unsigned short* hbuf  = (unsigned short*)carve((size_t)NTOK * H_ * 2);    // reused per rank

  init_counts<<<1, 32, 0, stream>>>(cnt);
  router_kernel<<<NTOK / 8, 256, 0, stream>>>(x, router_w, router_b, cnt, tokL, gateL);
  cvt_f32_bf16<<<(NTOK * D_) / (256 * 8), 256, 0, stream>>>(x, xbf);
  transpose_cvt<<<dim3(H_ / 32, D_ / 32, E_), dim3(32, 8), 0, stream>>>(w1, w1t, D_, H_);
  transpose_cvt<<<dim3(D_ / 32, H_ / 32, E_), dim3(32, 8), 0, stream>>>(w2, w2t, H_, D_);

  for (int rank = 0; rank < 2; ++rank) {
    const int*   cR = cnt + rank * E_;
    const int*   tR = tokL + (size_t)rank * E_ * CAP;
    const float* gR = gateL + (size_t)rank * E_ * CAP;
    // FFN1: h = relu(x @ w1 + b1)
    moe_ffn_gemm<0><<<dim3(CAP / 128, H_ / 128, E_), 256, 0, stream>>>(
        xbf, w1t, b1, cR, tR, gR, hbuf, nullptr, D_, H_);
    // FFN2: out (rank0: =, rank1: +=) gate * (h @ w2 + b2)
    if (rank == 0)
      moe_ffn_gemm<1><<<dim3(CAP / 128, D_ / 128, E_), 256, 0, stream>>>(
          hbuf, w2t, b2, cR, tR, gR, nullptr, out, H_, D_);
    else
      moe_ffn_gemm<2><<<dim3(CAP / 128, D_ / 128, E_), 256, 0, stream>>>(
          hbuf, w2t, b2, cR, tR, gR, nullptr, out, H_, D_);
  }
}
